// GraphConv_1168231104947
// MI455X (gfx1250) — matrix operand order repaired and checked by reading, not compiled, not run on hardware
//
#include <hip/hip_runtime.h>
#include <hip/hip_bf16.h>
#include <math.h>

// Problem constants (match reference)
static constexpr int NV = 40000;
static constexpr int NC = 300;
static constexpr int NI = 7000;
static constexpr int NN = NV + NC + NI;          // 47300
static constexpr int D  = 128;
static constexpr int MPAD = ((NN + 15) / 16) * 16; // 47312, multiple of 16 for WMMA tiles
static constexpr int MTILES = MPAD / 16;           // 2957

typedef __bf16 bf16_t;
typedef __attribute__((ext_vector_type(16))) __bf16 bf16x16;
typedef __attribute__((ext_vector_type(8)))  __bf16 bf16x8;
typedef __attribute__((ext_vector_type(8)))  float  f32x8;

// ---------------------------------------------------------------------------
// Float atomic max via sign-aware integer atomics (lowers to
// global_atomic_max_i32 / global_atomic_min_u32 on gfx1250).
__device__ __forceinline__ void atomicMaxF(float* addr, float val) {
    if (val >= 0.0f) {
        atomicMax((int*)addr, __float_as_int(val));
    } else {
        atomicMin((unsigned int*)addr, __float_as_uint(val));
    }
}

// ---------------------------------------------------------------------------
// Pack: build concatenated center (f32 + bf16, padded rows zeroed) and
// relu'd offset tables.
__global__ void pack_nodes(const float* __restrict__ vc, const float* __restrict__ vo,
                           const float* __restrict__ cc, const float* __restrict__ co,
                           const float* __restrict__ ic, const float* __restrict__ io,
                           float* __restrict__ allC, float* __restrict__ allOff,
                           bf16_t* __restrict__ allCh) {
    int idx = blockIdx.x * 256 + threadIdx.x;
    if (idx >= MPAD * D) return;
    int row = idx >> 7;
    int col = idx & (D - 1);
    float c = 0.0f, o = 0.0f;
    if (row < NV) {
        c = vc[idx];
        o = vo[idx];
    } else if (row < NV + NC) {
        int r = row - NV;
        c = cc[r * D + col];
        o = co[r * D + col];
    } else if (row < NN) {
        int r = row - NV - NC;
        c = ic[r * D + col];
        o = io[r * D + col];
    }
    allCh[idx] = (bf16_t)c;
    if (row < NN) {
        allC[idx]   = c;
        allOff[idx] = fmaxf(o, 0.0f);   // all_offset = relu(concat(...))
    }
}

// Convert both attention weight matrices to bf16 (row-major [n][k], n=out dim).
__global__ void pack_weights(const float* __restrict__ w1, const float* __restrict__ w2,
                             bf16_t* __restrict__ w1h, bf16_t* __restrict__ w2h) {
    int idx = blockIdx.x * 256 + threadIdx.x;
    if (idx < D * D)          w1h[idx]         = (bf16_t)w1[idx];
    else if (idx < 2 * D * D) w2h[idx - D * D] = (bf16_t)w2[idx - D * D];
}

// Init scatter buffers: m = -FLT_MAX, den = 0, num = 0, out_offset = 0.
__global__ void init_scatter(float* __restrict__ m, float* __restrict__ den,
                             float* __restrict__ num, float* __restrict__ outOff) {
    int idx = blockIdx.x * 256 + threadIdx.x;
    if (idx >= NV * D) return;
    m[idx]      = -3.402823466e38f;
    den[idx]    = 0.0f;
    num[idx]    = 0.0f;
    outOff[idx] = 0.0f;
}

// ---------------------------------------------------------------------------
// Node-level MLP GEMM: Y[M,128] = act(X[M,128] @ W^T + b).
// One 16x16 output tile per wave, 8 waves (= 8 column tiles) per block,
// blockIdx.x = row tile. K = 128 -> 4 x v_wmma_f32_16x16x32_bf16.
//
// Fragment layouts per CDNA5 ISA 7.12.2:
//   A (16x32 MxK bf16): lane<16 -> row=lane, K = {k0..k0+7, k0+16..k0+23}
//                       lane>=16 -> row=lane-16, K shifted by +8.
//   B (32x16 KxN bf16): lane<16 -> col=lane,  K = {k0..k0+7, k0+16..k0+23}
//                       lane>=16 -> col=lane-16, K shifted by +8 (same order as A).
//   Since B[k][n] = W[n][k], each lane reads two 8-element runs of row n of W.
//   C/D (16x16 f32): vgpr r holds row (lane<16 ? r : r+8), col = lane&15.
template <bool RELU, bool OUT_BF16>
__global__ void mlp_gemm128(const bf16_t* __restrict__ X, const bf16_t* __restrict__ W,
                            const float* __restrict__ bias,
                            float* __restrict__ Yf, bf16_t* __restrict__ Yh) {
    int wave = threadIdx.x >> 5;         // 0..7 -> column tile
    int lane = threadIdx.x & 31;
    int tm = blockIdx.x;                 // row tile (0..MTILES-1)
    int tn = wave;
    int halfsel = lane >> 4;             // 0 or 1
    int row = tm * 16 + (lane & 15);     // A row for this lane
    int n   = tn * 16 + (lane & 15);     // B column (= output column) for this lane
    int kbA = halfsel * 8;
    int kbB = halfsel * 16;

    f32x8 acc = {};
#pragma unroll
    for (int kc = 0; kc < 4; ++kc) {
        int k0 = kc * 32;
        bf16x8 a0 = *(const bf16x8*)(X + row * D + k0 + kbA);
        bf16x8 a1 = *(const bf16x8*)(X + row * D + k0 + 16 + kbA);
        bf16x8 b0 = *(const bf16x8*)(W + n * D + k0 + kbA);
        bf16x8 b1 = *(const bf16x8*)(W + n * D + k0 + 16 + kbA);
        bf16x16 A = __builtin_shufflevector(a0, a1, 0,1,2,3,4,5,6,7,8,9,10,11,12,13,14,15);
        bf16x16 B = __builtin_shufflevector(b0, b1, 0,1,2,3,4,5,6,7,8,9,10,11,12,13,14,15);
        acc = __builtin_amdgcn_wmma_f32_16x16x32_bf16(
            /*neg_a=*/false, A, /*neg_b=*/false, B,
            /*c_mod=*/(short)0, acc, /*reuse_a=*/false, /*reuse_b=*/false);
    }

    float bcol = bias[tn * 16 + (lane & 15)];
#pragma unroll
    for (int r = 0; r < 8; ++r) {
        int m = tm * 16 + (halfsel ? (r + 8) : r);
        int c = tn * 16 + (lane & 15);
        float v = acc[r] + bcol;
        if (RELU) v = fmaxf(v, 0.0f);
        if (OUT_BF16) Yh[m * D + c] = (bf16_t)v;
        else          Yf[m * D + c] = v;
    }
}

// ---------------------------------------------------------------------------
// Edge pass 1: per-dim segment max of node logits.  One wave per edge,
// 4 dims per lane (float4 coalesced).
__global__ void edge_seg_max(const int* __restrict__ g, int E,
                             const float* __restrict__ L, float* __restrict__ m) {
    int e = blockIdx.x * 8 + (threadIdx.x >> 5);
    if (e >= E) return;
    int h = g[e];
    if ((unsigned)h >= (unsigned)NV) return;   // only segments [0,NV) are consumed
    int t = g[E + e];
    int lane = threadIdx.x & 31;
    float4 lv = *(const float4*)(L + (size_t)t * D + lane * 4);
    float* md = m + (size_t)h * D + lane * 4;
    atomicMaxF(md + 0, lv.x);
    atomicMaxF(md + 1, lv.y);
    atomicMaxF(md + 2, lv.z);
    atomicMaxF(md + 3, lv.w);
}

// Edge pass 2: e = exp(L[t]-m[h]); den[h] += e; num[h] += e * emb[t].
__global__ void edge_seg_acc(const int* __restrict__ g, int E,
                             const float* __restrict__ L, const float* __restrict__ m,
                             const float* __restrict__ allC,
                             float* __restrict__ den, float* __restrict__ num) {
    int e = blockIdx.x * 8 + (threadIdx.x >> 5);
    if (e >= E) return;
    int h = g[e];
    if ((unsigned)h >= (unsigned)NV) return;
    int t = g[E + e];
    int lane = threadIdx.x & 31;
    float4 lv = *(const float4*)(L    + (size_t)t * D + lane * 4);
    float4 mv = *(const float4*)(m    + (size_t)h * D + lane * 4);
    float4 cv = *(const float4*)(allC + (size_t)t * D + lane * 4);
    float e0 = __expf(lv.x - mv.x);
    float e1 = __expf(lv.y - mv.y);
    float e2 = __expf(lv.z - mv.z);
    float e3 = __expf(lv.w - mv.w);
    float* dd = den + (size_t)h * D + lane * 4;
    float* nd = num + (size_t)h * D + lane * 4;
    atomicAdd(dd + 0, e0); atomicAdd(nd + 0, e0 * cv.x);
    atomicAdd(dd + 1, e1); atomicAdd(nd + 1, e1 * cv.y);
    atomicAdd(dd + 2, e2); atomicAdd(nd + 2, e2 * cv.z);
    atomicAdd(dd + 3, e3); atomicAdd(nd + 3, e3 * cv.w);
}

// Finalize: agg = num / max(den, has-edges?den:1); out_emb = l2norm(agg).
// One wave per visit row; lane-shuffle reduction for the row norm.
__global__ void finalize_emb(const float* __restrict__ num, const float* __restrict__ den,
                             float* __restrict__ outEmb) {
    int v = blockIdx.x * 8 + (threadIdx.x >> 5);
    if (v >= NV) return;
    int lane = threadIdx.x & 31;
    float4 nu = *(const float4*)(num + (size_t)v * D + lane * 4);
    float4 de = *(const float4*)(den + (size_t)v * D + lane * 4);
    float a0 = nu.x / (de.x > 0.0f ? de.x : 1.0f);
    float a1 = nu.y / (de.y > 0.0f ? de.y : 1.0f);
    float a2 = nu.z / (de.z > 0.0f ? de.z : 1.0f);
    float a3 = nu.w / (de.w > 0.0f ? de.w : 1.0f);
    float ss = a0 * a0 + a1 * a1 + a2 * a2 + a3 * a3;
#pragma unroll
    for (int off = 16; off > 0; off >>= 1) ss += __shfl_xor(ss, off, 32);
    float inv = 1.0f / fmaxf(sqrtf(ss), 1e-12f);
    float4 r = make_float4(a0 * inv, a1 * inv, a2 * inv, a3 * inv);
    *(float4*)(outEmb + (size_t)v * D + lane * 4) = r;
}

// Offset scatter-max.  REQUIRE_T_GE_NV=true for graph1 (only iv/ut blocks);
// false for graph2 (iv2/ut2/vv all reduce to h<NV).  Output pre-initialized
// to 0, which implements both empty-segment behavior and the final relu/max0.
template <bool REQUIRE_T_GE_NV>
__global__ void edge_off_max(const int* __restrict__ g, int E,
                             const float* __restrict__ allOff, float* __restrict__ outOff) {
    int e = blockIdx.x * 8 + (threadIdx.x >> 5);
    if (e >= E) return;
    int h = g[e];
    if ((unsigned)h >= (unsigned)NV) return;
    int t = g[E + e];
    if (REQUIRE_T_GE_NV && t < NV) return;
    int lane = threadIdx.x & 31;
    float4 ov = *(const float4*)(allOff + (size_t)t * D + lane * 4);
    float* od = outOff + (size_t)h * D + lane * 4;
    atomicMaxF(od + 0, ov.x);
    atomicMaxF(od + 1, ov.y);
    atomicMaxF(od + 2, ov.z);
    atomicMaxF(od + 3, ov.w);
}

// ---------------------------------------------------------------------------
extern "C" void kernel_launch(void* const* d_in, const int* in_sizes, int n_in,
                              void* d_out, int out_size, void* d_ws, size_t ws_size,
                              hipStream_t stream) {
    const float* vc  = (const float*)d_in[0];
    const float* vo  = (const float*)d_in[1];
    const float* cc  = (const float*)d_in[2];
    const float* co  = (const float*)d_in[3];
    const float* ic  = (const float*)d_in[4];
    const float* io  = (const float*)d_in[5];
    // d_in[6] visit_time, d_in[11..14] tw1/tb1/tw2/tb2: dead (lam == 1.0).
    const float* aw1 = (const float*)d_in[7];
    const float* ab1 = (const float*)d_in[8];
    const float* aw2 = (const float*)d_in[9];
    const float* ab2 = (const float*)d_in[10];
    const int*   g1  = (const int*)d_in[15];
    const int*   g2  = (const int*)d_in[16];
    const int E1 = in_sizes[15] / 2;
    const int E2 = in_sizes[16] / 2;

    float* outEmb = (float*)d_out;                     // [NV, 128]
    float* outOff = (float*)d_out + (size_t)NV * D;    // [NV, 128]

    // Workspace carve (256B aligned)
    char* p = (char*)d_ws;
    auto carve = [&](size_t bytes) {
        void* r = (void*)p;
        p += (bytes + 255) & ~(size_t)255;
        return r;
    };
    float*  allC   = (float*)carve((size_t)NN * D * sizeof(float));
    float*  allOff = (float*)carve((size_t)NN * D * sizeof(float));
    float*  Lg     = (float*)carve((size_t)MPAD * D * sizeof(float));
    float*  segM   = (float*)carve((size_t)NV * D * sizeof(float));
    float*  segDen = (float*)carve((size_t)NV * D * sizeof(float));
    float*  segNum = (float*)carve((size_t)NV * D * sizeof(float));
    bf16_t* allCh  = (bf16_t*)carve((size_t)MPAD * D * sizeof(bf16_t));
    bf16_t* Hh     = (bf16_t*)carve((size_t)MPAD * D * sizeof(bf16_t));
    bf16_t* w1h    = (bf16_t*)carve((size_t)D * D * sizeof(bf16_t));
    bf16_t* w2h    = (bf16_t*)carve((size_t)D * D * sizeof(bf16_t));
    (void)ws_size; (void)n_in; (void)out_size;

    // 1) Pack nodes + weights, init scatter buffers.
    pack_nodes<<<(MPAD * D + 255) / 256, 256, 0, stream>>>(vc, vo, cc, co, ic, io,
                                                           allC, allOff, allCh);
    pack_weights<<<(2 * D * D + 255) / 256, 256, 0, stream>>>(aw1, aw2, w1h, w2h);
    init_scatter<<<(NV * D + 255) / 256, 256, 0, stream>>>(segM, segDen, segNum, outOff);

    // 2) Node-level attention MLP (hoisted out of the edge loop):
    //    H = relu(allC @ W1^T + b1)  [bf16 out];  L = H @ W2^T + b2  [f32 out].
    mlp_gemm128<true,  true ><<<MTILES, 256, 0, stream>>>(allCh, w1h, ab1, nullptr, Hh);
    mlp_gemm128<false, false><<<MTILES, 256, 0, stream>>>(Hh, w2h, ab2, Lg, nullptr);

    // 3) Segment softmax + weighted sum over graph1 edges (agg1 == final emb,
    //    since lam == 1.0 kills the agg2 branch exactly).
    edge_seg_max<<<(E1 + 7) / 8, 256, 0, stream>>>(g1, E1, Lg, segM);
    edge_seg_acc<<<(E1 + 7) / 8, 256, 0, stream>>>(g1, E1, Lg, segM, allC, segDen, segNum);
    finalize_emb<<<(NV + 7) / 8, 256, 0, stream>>>(segNum, segDen, outEmb);

    // 4) Offset blocks: max of 5 scatter-max blocks == two fused passes.
    edge_off_max<true ><<<(E1 + 7) / 8, 256, 0, stream>>>(g1, E1, allOff, outOff);
    edge_off_max<false><<<(E2 + 7) / 8, 256, 0, stream>>>(g2, E2, allOff, outOff);
}
